// GraphAttentionTransformerHAM_71554155151867
// MI455X (gfx1250) — matrix operand order repaired and checked by reading, not compiled, not run on hardware
//
#include <hip/hip_runtime.h>
#include <hip/hip_bf16.h>
#include <math.h>

// ---------------------------------------------------------------------------
// GraphAttentionTransformer for MI455X (gfx1250, wave32, WMMA).
// All dense algebra via v_wmma_f32_16x16x32_f16. Weights pre-packed to f16 in
// B-fragment order (contiguous 32B per lane per tile). A operands load as
// aligned float4 runs with fused gather / +e. Epilogue fuses bias / SiLU /
// residual. Dual-output kernel computes k and v from one shared A read.
// ---------------------------------------------------------------------------

typedef __attribute__((ext_vector_type(16))) _Float16 v16h;
typedef __attribute__((ext_vector_type(8)))  float    v8f;

__device__ __forceinline__ int f2ord(float f) {
  int i = __float_as_int(f);
  return (i >= 0) ? i : (i ^ 0x7FFFFFFF);
}
__device__ __forceinline__ float ord2f(int i) {
  return __int_as_float((i >= 0) ? i : (i ^ 0x7FFFFFFF));
}
__device__ __forceinline__ float silu_f(float v) {
  return v / (1.0f + __expf(-v));
}

// ---------------------------------------------------------------------------
// Weight pack: f32 [Ksrc,Ncsrc] row-major -> f16 B-fragment order, padded to
// [Kpad,Ncpad].  Block per (kchunk, ntile): 32 lanes x 16 halfs; lane L holds
// col tn*16+(L&15), K = kc*32+(L>>4)*8+(h&7)+(h>>3)*16.
// ---------------------------------------------------------------------------
__global__ void pack_w(const float* __restrict__ W, _Float16* __restrict__ Wp,
                       int Ksrc, int Kpad, int Ncsrc, int Ncpad)
{
  const int t = blockIdx.x * blockDim.x + threadIdx.x;
  const int tilesN = Ncpad >> 4;
  const int total = (Kpad >> 5) * tilesN * 32;
  if (t >= total) return;
  const int lane = t & 31;
  const int rest = t >> 5;                 // kc*tilesN + tn
  const int tn = rest % tilesN;
  const int kc = rest / tilesN;
  const int col = tn * 16 + (lane & 15);
  const int kb  = kc * 32 + ((lane >> 4) << 3);
  _Float16* dst = Wp + ((size_t)rest * 32 + lane) * 16;
#pragma unroll
  for (int h = 0; h < 16; ++h) {
    const int k = kb + (h & 7) + ((h >> 3) << 4);
    float v = (k < Ksrc && col < Ncsrc) ? W[(size_t)k * Ncsrc + col] : 0.0f;
    dst[h] = (_Float16)v;
  }
}

// ---------------------------------------------------------------------------
// A-fragment loader: two contiguous float4 pairs per lane (CDNA5 16-bit A
// 16x32 layout), optional fused +Eadd, converted to f16.
// ---------------------------------------------------------------------------
template<bool HAS_EADD>
__device__ __forceinline__ v16h load_a_frag(const float* ap, const float* ep) {
  float4 p0 = *(const float4*)(ap);
  float4 p1 = *(const float4*)(ap + 4);
  float4 p2 = *(const float4*)(ap + 16);
  float4 p3 = *(const float4*)(ap + 20);
  if (HAS_EADD) {
    float4 q0 = *(const float4*)(ep);
    float4 q1 = *(const float4*)(ep + 4);
    float4 q2 = *(const float4*)(ep + 16);
    float4 q3 = *(const float4*)(ep + 20);
    p0.x += q0.x; p0.y += q0.y; p0.z += q0.z; p0.w += q0.w;
    p1.x += q1.x; p1.y += q1.y; p1.z += q1.z; p1.w += q1.w;
    p2.x += q2.x; p2.y += q2.y; p2.z += q2.z; p2.w += q2.w;
    p3.x += q3.x; p3.y += q3.y; p3.z += q3.z; p3.w += q3.w;
  }
  v16h a;
  a[0]  = (_Float16)p0.x; a[1]  = (_Float16)p0.y;
  a[2]  = (_Float16)p0.z; a[3]  = (_Float16)p0.w;
  a[4]  = (_Float16)p1.x; a[5]  = (_Float16)p1.y;
  a[6]  = (_Float16)p1.z; a[7]  = (_Float16)p1.w;
  a[8]  = (_Float16)p2.x; a[9]  = (_Float16)p2.y;
  a[10] = (_Float16)p2.z; a[11] = (_Float16)p2.w;
  a[12] = (_Float16)p3.x; a[13] = (_Float16)p3.y;
  a[14] = (_Float16)p3.z; a[15] = (_Float16)p3.w;
  return a;
}

// ---------------------------------------------------------------------------
// Fast WMMA GEMM. Y = (X[g(row),:] (+Eadd)) @ W  (+bias)(silu)(+Cadd)
// K%32==0 (row stride == K), Ncols%(16*TN)==0, M%16==0.
// ---------------------------------------------------------------------------
template<int TN, bool HAS_IDX, bool HAS_EADD, bool HAS_CADD, bool HAS_BIAS, bool DO_SILU>
__global__ __launch_bounds__(256)
void wmma_gemm_fast(const float* __restrict__ X, const int* __restrict__ idx,
                    const float* __restrict__ Eadd, const _Float16* __restrict__ Wp,
                    const float* __restrict__ bias, const float* __restrict__ Cadd,
                    float* __restrict__ Y, int M, int K, int Ncols)
{
  const int lane    = threadIdx.x & 31;
  const int wave    = (blockIdx.x * blockDim.x + threadIdx.x) >> 5;
  const int tilesN  = Ncols >> 4;
  const int groupsN = tilesN / TN;
  const int tm = wave / groupsN;
  const int gn = wave - tm * groupsN;
  if (tm >= (M >> 4)) return;              // wave-uniform exit

  const int rl = lane & 15;
  const int hi = lane >> 4;
  const int row = tm * 16 + rl;
  const int g = HAS_IDX ? idx[row] : row;
  const float* xrow = X + (size_t)g * K;
  const float* erow = HAS_EADD ? (Eadd + (size_t)row * K) : nullptr;
  const int kbaseA = hi << 3;              // 0 or 8

  v8f acc[TN] = {};
  for (int k0 = 0; k0 < K; k0 += 32) {
    v16h a = load_a_frag<HAS_EADD>(xrow + k0 + kbaseA,
                                   HAS_EADD ? (erow + k0 + kbaseA) : nullptr);
    const _Float16* bp =
        Wp + ((size_t)((k0 >> 5) * tilesN + gn * TN) * 32 + lane) * 16;
#pragma unroll
    for (int j = 0; j < TN; ++j) {
      v16h b = *(const v16h*)(bp + (size_t)j * 512);
      acc[j] = __builtin_amdgcn_wmma_f32_16x16x32_f16(false, a, false, b,
                                                      (short)0, acc[j], false, false);
    }
  }

  const int mrow0 = tm * 16 + (hi << 3);
#pragma unroll
  for (int j = 0; j < TN; ++j) {
    const int col = (gn * TN + j) * 16 + rl;
#pragma unroll
    for (int r = 0; r < 8; ++r) {
      const size_t off = (size_t)(mrow0 + r) * Ncols + col;
      float v = acc[j][r];
      if (HAS_BIAS) v += bias[col];
      if (DO_SILU)  v = silu_f(v);
      if (HAS_CADD) v += Cadd[off];
      Y[off] = v;
    }
  }
}

// ---------------------------------------------------------------------------
// Dual-output WMMA GEMM for k,v:  A = h[edge_src] + e  read ONCE, multiplied
// by two packed weights.  Fixed TN=4, Ncols=64.
// ---------------------------------------------------------------------------
__global__ __launch_bounds__(256)
void wmma_gemm_dual(const float* __restrict__ X, const int* __restrict__ idx,
                    const float* __restrict__ Eadd,
                    const _Float16* __restrict__ Wp0, const _Float16* __restrict__ Wp1,
                    float* __restrict__ Y0, float* __restrict__ Y1, int M, int K)
{
  const int lane = threadIdx.x & 31;
  const int tm   = (blockIdx.x * blockDim.x + threadIdx.x) >> 5;
  if (tm >= (M >> 4)) return;

  const int rl = lane & 15;
  const int hi = lane >> 4;
  const int row = tm * 16 + rl;
  const int g = idx[row];
  const float* xrow = X + (size_t)g * K;
  const float* erow = Eadd + (size_t)row * K;
  const int kbaseA = hi << 3;

  v8f acc0[4] = {}, acc1[4] = {};
  for (int k0 = 0; k0 < K; k0 += 32) {
    v16h a = load_a_frag<true>(xrow + k0 + kbaseA, erow + k0 + kbaseA);
    const size_t boff = ((size_t)((k0 >> 5) * 4) * 32 + lane) * 16;
#pragma unroll
    for (int j = 0; j < 4; ++j) {
      v16h b = *(const v16h*)(Wp0 + boff + (size_t)j * 512);
      acc0[j] = __builtin_amdgcn_wmma_f32_16x16x32_f16(false, a, false, b,
                                                       (short)0, acc0[j], false, false);
    }
#pragma unroll
    for (int j = 0; j < 4; ++j) {
      v16h b = *(const v16h*)(Wp1 + boff + (size_t)j * 512);
      acc1[j] = __builtin_amdgcn_wmma_f32_16x16x32_f16(false, a, false, b,
                                                       (short)0, acc1[j], false, false);
    }
  }

  const int mrow0 = tm * 16 + (hi << 3);
#pragma unroll
  for (int j = 0; j < 4; ++j) {
    const int col = j * 16 + rl;
#pragma unroll
    for (int r = 0; r < 8; ++r) {
      const size_t off = (size_t)(mrow0 + r) * 64 + col;
      Y0[off] = acc0[j][r];
      Y1[off] = acc1[j][r];
    }
  }
}

// ---------------------------------------------------------------------------
// LayerNorm (no affine), one wave per row, wave32 shfl reductions.
// ---------------------------------------------------------------------------
__global__ void ln_kernel(const float* __restrict__ in, float* __restrict__ out,
                          int M, int D, int doRelu)
{
  const int lane = threadIdx.x & 31;
  const int row  = (blockIdx.x * blockDim.x + threadIdx.x) >> 5;
  if (row >= M) return;
  const float* p = in + (size_t)row * D;
  float s = 0.0f;
  for (int j = lane; j < D; j += 32) s += p[j];
#pragma unroll
  for (int off = 16; off; off >>= 1) s += __shfl_xor(s, off);
  const float mean = s / (float)D;
  float v = 0.0f;
  for (int j = lane; j < D; j += 32) { float d = p[j] - mean; v += d * d; }
#pragma unroll
  for (int off = 16; off; off >>= 1) v += __shfl_xor(v, off);
  const float inv = rsqrtf(v / (float)D + 1e-6f);
  float* q = out + (size_t)row * D;
  for (int j = lane; j < D; j += 32) {
    float y = (p[j] - mean) * inv;
    if (doRelu) y = fmaxf(y, 0.0f);
    q[j] = y;
  }
}

// ---------------------------------------------------------------------------
// Edge featurization: r, SH(l<=2), RBF, 160-wide (137 + zero pad) context
// row [nf_src(32) | nf_dst(32) | sh(9) | rbf(64) | 0 pad(23)].
// ---------------------------------------------------------------------------
__global__ void edge_prep(const int* __restrict__ esrc, const int* __restrict__ edst,
                          const float* __restrict__ evec, const int* __restrict__ atom,
                          const float* __restrict__ emb,   // [120,32]
                          float* __restrict__ rbf,         // [E,64]
                          float* __restrict__ ectx,        // [E,160]
                          float* __restrict__ rout,        // [E]
                          int E)
{
  int e = blockIdx.x * blockDim.x + threadIdx.x;
  if (e >= E) return;
  const float vx = evec[e * 3 + 0], vy = evec[e * 3 + 1], vz = evec[e * 3 + 2];
  const float r = sqrtf(vx * vx + vy * vy + vz * vz);
  rout[e] = r;
  const float inv = 1.0f / (r + 1e-12f);
  const float x = vy * inv, y = vz * inv, z = vx * inv;   // u = vec[[1,2,0]]/r
  const float s3 = 1.7320508075688772f, s15 = 3.872983346207417f, s5 = 2.23606797749979f;
  float sh[9];
  sh[0] = 1.0f;            sh[1] = s3 * x;          sh[2] = s3 * y;
  sh[3] = s3 * z;          sh[4] = s15 * x * y;     sh[5] = s15 * y * z;
  sh[6] = 0.5f * s5 * (3.0f * z * z - 1.0f);
  sh[7] = s15 * x * z;     sh[8] = 0.5f * s15 * (x * x - y * y);

  float* ctx = ectx + (size_t)e * 160;
  const float* ns = emb + (size_t)atom[esrc[e]] * 32;
  const float* nd = emb + (size_t)atom[edst[e]] * 32;
#pragma unroll
  for (int j = 0; j < 32; ++j) ctx[j]      = ns[j];
#pragma unroll
  for (int j = 0; j < 32; ++j) ctx[32 + j] = nd[j];
#pragma unroll
  for (int j = 0; j < 9; ++j)  ctx[64 + j] = sh[j];

  const float w = 0.5f * 10.0f / 64.0f;          // 0.078125
  const float den = 1.0f / (2.0f * w * w);
  float* rb = rbf + (size_t)e * 64;
#pragma unroll 8
  for (int j = 0; j < 64; ++j) {
    float c = 10.0f * (float)j / 63.0f;
    float d = r - c;
    float val = __expf(-d * d * den);
    rb[j] = val;
    ctx[73 + j] = val;
  }
#pragma unroll
  for (int j = 137; j < 160; ++j) ctx[j] = 0.0f;
}

// ---------------------------------------------------------------------------
// Attention: logits + segment max / exp-sum / weighted aggregation.
// radial has row stride 16 (padded N from the Wr2 GEMM).
// ---------------------------------------------------------------------------
__global__ void logits_kernel(const float* __restrict__ q, const float* __restrict__ k,
                              const float* __restrict__ radial, const float* __restrict__ r,
                              const int* __restrict__ edst,
                              float* __restrict__ logits, int* __restrict__ segmax, int E)
{
  int t = blockIdx.x * blockDim.x + threadIdx.x;
  if (t >= E * 4) return;
  const int e = t >> 2, h = t & 3;
  const float* qe = q + (size_t)e * 64 + h * 16;
  const float* ke = k + (size_t)e * 64 + h * 16;
  float acc = 0.0f;
#pragma unroll
  for (int d = 0; d < 16; ++d) acc += qe[d] * ke[d];
  float lg = acc * 0.25f + radial[(size_t)e * 16 + h];      // 1/sqrt(16)
  if (!(r[e] < 10.0f)) lg = -1e9f;
  logits[t] = lg;
  atomicMax(&segmax[edst[e] * 4 + h], f2ord(lg));
}

__global__ void expsum_kernel(float* __restrict__ logits, const int* __restrict__ segmax,
                              float* __restrict__ segsum, const int* __restrict__ edst, int E)
{
  int t = blockIdx.x * blockDim.x + threadIdx.x;
  if (t >= E * 4) return;
  const int e = t >> 2, h = t & 3;
  const float mx = ord2f(segmax[edst[e] * 4 + h]);
  const float ex = __expf(logits[t] - mx);
  logits[t] = ex;
  atomicAdd(&segsum[edst[e] * 4 + h], ex);
}

__global__ void msg_kernel(const float* __restrict__ ex, const float* __restrict__ segsum,
                           const float* __restrict__ v, const int* __restrict__ edst,
                           float* __restrict__ agg, int E)
{
  long long t = (long long)blockIdx.x * blockDim.x + threadIdx.x;
  if (t >= (long long)E * 64) return;
  const int e = (int)(t >> 6), d = (int)(t & 63), h = d >> 4;
  const int dst = edst[e];
  const float alpha = ex[(size_t)e * 4 + h] / (segsum[dst * 4 + h] + 1e-9f);
  atomicAdd(&agg[(size_t)dst * 64 + d], alpha * v[(size_t)e * 64 + d]);
}

// ---------------------------------------------------------------------------
// Fill / pool.  energy has row stride 16 (padded N=1 GEMM).
// ---------------------------------------------------------------------------
__global__ void fillf_kernel(float* __restrict__ p, float val, long long n) {
  long long i = (long long)blockIdx.x * blockDim.x + threadIdx.x;
  if (i < n) p[i] = val;
}
__global__ void filli_kernel(int* __restrict__ p, int val, long long n) {
  long long i = (long long)blockIdx.x * blockDim.x + threadIdx.x;
  if (i < n) p[i] = val;
}
__global__ void pool_kernel(const float* __restrict__ energy, const int* __restrict__ batch,
                            const float* __restrict__ hb3, float* __restrict__ out, int N) {
  int n = blockIdx.x * blockDim.x + threadIdx.x;
  if (n >= N) return;
  atomicAdd(&out[batch[n]], (energy[(size_t)n * 16] + hb3[0]) * 0.11785113019775793f);
}

// ---------------------------------------------------------------------------
// Host orchestration.
// ---------------------------------------------------------------------------
static inline unsigned cdivu(long long a, long long b) { return (unsigned)((a + b - 1) / b); }
static inline size_t pk_size(int Kpad, int Ncpad) {       // halfs
  return (size_t)(Kpad / 32) * (Ncpad / 16) * 512;
}

extern "C" void kernel_launch(void* const* d_in, const int* in_sizes, int n_in,
                              void* d_out, int out_size, void* d_ws, size_t ws_size,
                              hipStream_t stream) {
  const int* edge_src  = (const int*)d_in[1];
  const int* edge_dst  = (const int*)d_in[2];
  const float* edge_vec = (const float*)d_in[3];
  const int* batch     = (const int*)d_in[4];
  const int* node_atom = (const int*)d_in[5];
  const float* ele_emb = (const float*)d_in[6];
  const float* W_embed = (const float*)d_in[7];
  const float* W_edge  = (const float*)d_in[8];
  const float* Wq = (const float*)d_in[9];
  const float* Wk = (const float*)d_in[10];
  const float* Wv = (const float*)d_in[11];
  const float* Wr1 = (const float*)d_in[12];
  const float* Wr2 = (const float*)d_in[13];
  const float* Wo  = (const float*)d_in[14];
  const float* W1  = (const float*)d_in[15];
  const float* W2  = (const float*)d_in[16];
  const float* W_feat = (const float*)d_in[17];
  const float* hW1 = (const float*)d_in[18];
  const float* hb1 = (const float*)d_in[19];
  const float* hW2 = (const float*)d_in[20];
  const float* hb2 = (const float*)d_in[21];
  const float* hW3 = (const float*)d_in[22];
  const float* hb3 = (const float*)d_in[23];

  const int E = in_sizes[1];
  const int N = in_sizes[4];
  const int L = 6;

  // ---- workspace carve-up (floats, then f16 pack arena) -----------------
  float* F = (float*)d_ws;
  size_t o = 0;
  auto alloc = [&](size_t n) { float* p = F + o; o += n; return p; };
  float* rbf    = alloc((size_t)E * 64);
  float* ectx   = alloc((size_t)E * 160);
  float* rE     = alloc((size_t)E);
  float* eEmb   = alloc((size_t)E * 64);
  float* x      = alloc((size_t)N * 64);
  float* h      = alloc((size_t)N * 64);
  float* qb     = alloc((size_t)E * 64);
  float* kb     = alloc((size_t)E * 64);
  float* vb     = alloc((size_t)E * 64);
  float* rad    = alloc((size_t)E * 64);
  float* radial = alloc((size_t)E * 16);
  float* logits = alloc((size_t)E * 4);
  int*   segmax = (int*)alloc((size_t)N * 4);
  float* segsum = alloc((size_t)N * 4);
  float* agg    = alloc((size_t)N * 64);
  float* ffn1   = alloc((size_t)N * 256);
  float* feat   = alloc((size_t)N * 512);
  float* featln = alloc((size_t)N * 512);
  float* t1     = alloc((size_t)N * 64);
  float* t2     = alloc((size_t)N * 64);
  float* energy = alloc((size_t)N * 16);

  _Float16* Harena = (_Float16*)(F + o);
  size_t ho = 0;
  auto halloc = [&](size_t n) { _Float16* p = Harena + ho; ho += n; return p; };
  _Float16* pWembed = halloc(pk_size(32, 64));
  _Float16* pWedge  = halloc(pk_size(160, 64));
  _Float16* pWfeat  = halloc(pk_size(64, 512));
  _Float16* pHW1    = halloc(pk_size(512, 64));
  _Float16* pHW2    = halloc(pk_size(64, 64));
  _Float16* pHW3    = halloc(pk_size(64, 16));
  _Float16 *pWq[6], *pWk[6], *pWv[6], *pWr1[6], *pWr2[6], *pWo[6], *pW1[6], *pW2[6];
  for (int l = 0; l < L; ++l) {
    pWq[l]  = halloc(pk_size(64, 64));
    pWk[l]  = halloc(pk_size(64, 64));
    pWv[l]  = halloc(pk_size(64, 64));
    pWr1[l] = halloc(pk_size(64, 64));
    pWr2[l] = halloc(pk_size(64, 16));
    pWo[l]  = halloc(pk_size(64, 64));
    pW1[l]  = halloc(pk_size(64, 256));
    pW2[l]  = halloc(pk_size(256, 64));
  }
  (void)ws_size;

  auto pack = [&](const float* W, _Float16* Wp, int Ks, int Kp, int Ns, int Np) {
    long long tot = (long long)(Kp / 32) * (Np / 16) * 32;
    pack_w<<<cdivu(tot, 256), 256, 0, stream>>>(W, Wp, Ks, Kp, Ns, Np);
  };
  auto ln = [&](const float* in, float* out, int M, int D, int relu) {
    ln_kernel<<<cdivu((long long)M * 32, 256), 256, 0, stream>>>(in, out, M, D, relu);
  };
  auto gblocks = [&](int M, int Nc, int TN) {
    long long waves = (long long)(M / 16) * ((Nc / 16) / TN);
    return cdivu(waves * 32, 256);
  };

  // ---- pack all weights to f16 B-fragment order -------------------------
  pack(W_embed, pWembed, 32, 32, 64, 64);
  pack(W_edge,  pWedge, 137, 160, 64, 64);
  pack(W_feat,  pWfeat,  64,  64, 512, 512);
  pack(hW1, pHW1, 512, 512, 64, 64);
  pack(hW2, pHW2,  64,  64, 64, 64);
  pack(hW3, pHW3,  64,  64,  1, 16);
  for (int l = 0; l < L; ++l) {
    pack(Wq  + (size_t)l * 4096,  pWq[l],  64, 64, 64, 64);
    pack(Wk  + (size_t)l * 4096,  pWk[l],  64, 64, 64, 64);
    pack(Wv  + (size_t)l * 4096,  pWv[l],  64, 64, 64, 64);
    pack(Wr1 + (size_t)l * 4096,  pWr1[l], 64, 64, 64, 64);
    pack(Wr2 + (size_t)l * 256,   pWr2[l], 64, 64, 4, 16);
    pack(Wo  + (size_t)l * 4096,  pWo[l],  64, 64, 64, 64);
    pack(W1  + (size_t)l * 16384, pW1[l],  64, 64, 256, 256);
    pack(W2  + (size_t)l * 16384, pW2[l], 256, 256, 64, 64);
  }

  // ---- edge features + embeddings ---------------------------------------
  edge_prep<<<cdivu(E, 256), 256, 0, stream>>>(edge_src, edge_dst, edge_vec, node_atom,
                                               ele_emb, rbf, ectx, rE, E);
  // x = ele_embed_w[node_atom] @ W_embed
  wmma_gemm_fast<4, true, false, false, false, false><<<gblocks(N, 64, 4), 256, 0, stream>>>(
      ele_emb, node_atom, nullptr, pWembed, nullptr, nullptr, x, N, 32, 64);
  // e = ectx @ W_edge   (K padded 137->160)
  wmma_gemm_fast<4, false, false, false, false, false><<<gblocks(E, 64, 4), 256, 0, stream>>>(
      ectx, nullptr, nullptr, pWedge, nullptr, nullptr, eEmb, E, 160, 64);

  // ---- transformer layers ------------------------------------------------
  for (int l = 0; l < L; ++l) {
    ln(x, h, N, 64, 0);
    // q = h[edge_dst] @ Wq
    wmma_gemm_fast<4, true, false, false, false, false><<<gblocks(E, 64, 4), 256, 0, stream>>>(
        h, edge_dst, nullptr, pWq[l], nullptr, nullptr, qb, E, 64, 64);
    // k,v = (h[edge_src] + e) @ {Wk, Wv}   -- one shared A read
    wmma_gemm_dual<<<cdivu((long long)(E / 16) * 32, 256), 256, 0, stream>>>(
        h, edge_src, eEmb, pWk[l], pWv[l], kb, vb, E, 64);
    // rad = silu(rbf @ Wr1)   (silu fused in epilogue)
    wmma_gemm_fast<4, false, false, false, false, true><<<gblocks(E, 64, 4), 256, 0, stream>>>(
        rbf, nullptr, nullptr, pWr1[l], nullptr, nullptr, rad, E, 64, 64);
    // radial = rad @ Wr2   (N padded 4->16)
    wmma_gemm_fast<1, false, false, false, false, false><<<gblocks(E, 16, 1), 256, 0, stream>>>(
        rad, nullptr, nullptr, pWr2[l], nullptr, nullptr, radial, E, 64, 16);

    filli_kernel<<<cdivu((long long)N * 4, 256), 256, 0, stream>>>(segmax, (int)0x80000000,
                                                                   (long long)N * 4);
    fillf_kernel<<<cdivu((long long)N * 4, 256), 256, 0, stream>>>(segsum, 0.0f, (long long)N * 4);
    fillf_kernel<<<cdivu((long long)N * 64, 256), 256, 0, stream>>>(agg, 0.0f, (long long)N * 64);

    logits_kernel<<<cdivu((long long)E * 4, 256), 256, 0, stream>>>(qb, kb, radial, rE, edge_dst,
                                                                    logits, segmax, E);
    expsum_kernel<<<cdivu((long long)E * 4, 256), 256, 0, stream>>>(logits, segmax, segsum,
                                                                    edge_dst, E);
    msg_kernel<<<cdivu((long long)E * 64, 256), 256, 0, stream>>>(logits, segsum, vb, edge_dst,
                                                                  agg, E);

    // x += agg @ Wo
    wmma_gemm_fast<4, false, false, true, false, false><<<gblocks(N, 64, 4), 256, 0, stream>>>(
        agg, nullptr, nullptr, pWo[l], nullptr, x, x, N, 64, 64);
    ln(x, h, N, 64, 0);
    // ffn1 = silu(h @ W1)   (silu fused)
    wmma_gemm_fast<4, false, false, false, false, true><<<gblocks(N, 256, 4), 256, 0, stream>>>(
        h, nullptr, nullptr, pW1[l], nullptr, nullptr, ffn1, N, 64, 256);
    // x += ffn1 @ W2
    wmma_gemm_fast<4, false, false, true, false, false><<<gblocks(N, 64, 4), 256, 0, stream>>>(
        ffn1, nullptr, nullptr, pW2[l], nullptr, x, x, N, 256, 64);
  }

  // ---- readout head ------------------------------------------------------
  wmma_gemm_fast<4, false, false, false, false, false><<<gblocks(N, 512, 4), 256, 0, stream>>>(
      x, nullptr, nullptr, pWfeat, nullptr, nullptr, feat, N, 64, 512);
  ln(feat, featln, N, 512, 0);
  wmma_gemm_fast<4, false, false, false, true, false><<<gblocks(N, 64, 4), 256, 0, stream>>>(
      featln, nullptr, nullptr, pHW1, hb1, nullptr, t1, N, 512, 64);
  ln(t1, t1, N, 64, 1);
  wmma_gemm_fast<4, false, false, false, true, false><<<gblocks(N, 64, 4), 256, 0, stream>>>(
      t1, nullptr, nullptr, pHW2, hb2, nullptr, t2, N, 64, 64);
  ln(t2, t2, N, 64, 1);
  wmma_gemm_fast<1, false, false, false, false, false><<<gblocks(N, 16, 1), 256, 0, stream>>>(
      t2, nullptr, nullptr, pHW3, nullptr, nullptr, energy, N, 64, 16);

  fillf_kernel<<<cdivu(out_size, 256), 256, 0, stream>>>((float*)d_out, 0.0f, (long long)out_size);
  pool_kernel<<<cdivu(N, 256), 256, 0, stream>>>(energy, batch, hb3, (float*)d_out, N);
}
